// LSTMDecoder_20486994001989
// MI455X (gfx1250) — matrix operand order repaired and checked by reading, not compiled, not run on hardware
//
#include <hip/hip_runtime.h>
#include <hip/hip_bf16.h>
#include <cstdint>

// ---------------------------------------------------------------------------
// LSTM decoder, MI455X (gfx1250): bf16 WMMA GEMMs w/ f32 accumulate, TDM
// (tensor_load_to_lds) staging of W_hh into LDS, double-buffered, shared by
// 8 waves per block.  B=128, E=512, H=1024, T=256, O=512 (fixed by reference).
// ---------------------------------------------------------------------------

typedef __attribute__((ext_vector_type(16))) __bf16       v16bf;
typedef __attribute__((ext_vector_type(8)))  float        v8f;
typedef __attribute__((ext_vector_type(4)))  unsigned int u32x4;
typedef __attribute__((ext_vector_type(8)))  int          i32x8;
typedef __attribute__((ext_vector_type(4)))  int          i32x4;

__device__ __forceinline__ float sigmoidf(float x) {
    return 1.0f / (1.0f + __expf(-x));
}
__device__ __forceinline__ float tanh_fast(float x) {
    return 1.0f - 2.0f / (__expf(2.0f * x) + 1.0f);
}

// A fragment: 16x32 bf16 tile, row-major, row stride = ld elements.
// ISA layout: lanes 0-15 hold M=0..15, K {0..7,16..23}; lanes 16-31 K {8..15,24..31}.
__device__ __forceinline__ v16bf load_a_frag(const __bf16* tile, int ld) {
    const int lane = threadIdx.x & 31;
    const int m    = lane & 15;
    const int kc   = (lane >> 4) << 3;               // 0 or 8
    const __bf16* p = tile + (long)m * ld + kc;
    union { v16bf v; uint4 u[2]; } t;
    t.u[0] = *reinterpret_cast<const uint4*>(p);      // K kc..kc+7
    t.u[1] = *reinterpret_cast<const uint4*>(p + 16); // K kc+16..kc+23
    return t.v;
}

// B fragment: 32x16 bf16, B[k][n] = W[n][k] with W row-major [N,K], row stride ld.
// Lanes 0-15: N=lane, K {0..7,16..23}; lanes 16-31: N=lane-16, K {8..15,24..31}.
__device__ __forceinline__ v16bf load_b_frag(const __bf16* wtile, int ld) {
    const int lane = threadIdx.x & 31;
    const int n    = lane & 15;
    const int k0   = (lane >> 4) << 3;               // 0 or 8
    const __bf16* p = wtile + (long)n * ld + k0;
    union { v16bf v; uint4 u[2]; } t;
    t.u[0] = *reinterpret_cast<const uint4*>(p);      // K k0..k0+7
    t.u[1] = *reinterpret_cast<const uint4*>(p + 16); // K k0+16..k0+23
    return t.v;
}

// ---------------------------------------------------------------------------
// TDM: DMA one 16-row x 128-col bf16 tile of W_hh (row stride 1024 elements)
// into contiguous LDS [16][128].  D# per CDNA5 ISA ch.8 (group0/group1).
// Toolchain here uses the 6-arg builtin: (g0, g1, g2, g3, g4, cpol).
// ---------------------------------------------------------------------------
__device__ __forceinline__ void tdm_load_tile_16x128(__bf16* lds_ptr, const __bf16* gptr) {
#if __has_builtin(__builtin_amdgcn_tensor_load_to_lds)
    const unsigned           lds = (unsigned)(uintptr_t)lds_ptr;      // low 32b = LDS byte addr
    const unsigned long long ga  = (unsigned long long)(uintptr_t)gptr;
    // group0: count=1 | lds_addr | global_addr[56:0] | type=2
    u32x4 g0 = { 1u, lds, (unsigned)ga, (unsigned)(ga >> 32) | 0x80000000u };
    // group1: data_size=1(2B); tensor_dim0=1024; tensor_dim1=4096;
    //         tile_dim0=128; tile_dim1=16; tile_dim2=0;
    //         tensor_dim0_stride=1024; tensor_dim1_stride=0
    i32x8 g1 = { (int)0x00010000u,          // mask=0, data_size=1
                 (int)(1024u << 16),        // tensor_dim0[15:0] @ bits 63:48
                 (int)(4096u << 16),        // dim0[31:16]=0 | tensor_dim1[15:0]
                 (int)(128u  << 16),        // dim1[31:16]=0 | tile_dim0
                 16,                        // tile_dim1 | tile_dim2=0
                 1024,                      // tensor_dim0_stride[31:0]
                 0, 0 };                    // stride0[47:32]=0 | stride1=0
    i32x4 z4 = { 0, 0, 0, 0 };
    i32x8 z8 = { 0, 0, 0, 0, 0, 0, 0, 0 };
    __builtin_amdgcn_tensor_load_to_lds(g0, g1, z4, z4, z8, 0);
#else
    // Fallback: wave-cooperative copy (32 lanes), 256 uint4 = 4KB.
    const int lane = threadIdx.x & 31;
#pragma unroll
    for (int i = 0; i < 8; ++i) {
        const int idx = i * 32 + lane;          // 0..255
        const int row = idx >> 4;               // /16
        const int c16 = idx & 15;               // uint4 within row
        reinterpret_cast<uint4*>(lds_ptr)[idx] =
            reinterpret_cast<const uint4*>(gptr + (long)row * 1024)[c16];
    }
#endif
}

// Generic C[M,N] = A[M,K]_bf16 @ W[N,K]_bf16^T + bias, f32 out with row stride ldc.
// One 16x16 tile per wave, 4 waves (128 threads) per block, grid sized exactly.
__global__ void wmma_gemm_bias(const __bf16* __restrict__ A,
                               const __bf16* __restrict__ W,
                               const float*  __restrict__ bias,
                               float* __restrict__ C,
                               int M, int N, int K, long ldc) {
    const int wave = threadIdx.x >> 5;
    const int tile = blockIdx.x * 4 + wave;
    const int tnc  = N >> 4;
    const int tm   = tile / tnc;
    const int tn   = tile - tm * tnc;
    if (tm >= (M >> 4)) return;                       // wave-uniform guard

    v8f acc = {};
    const __bf16* Arow = A + (long)tm * 16 * K;
    const __bf16* Wrow = W + (long)tn * 16 * K;
    for (int k = 0; k < K; k += 32) {
        v16bf a = load_a_frag(Arow + k, K);
        v16bf b = load_b_frag(Wrow + k, K);
        acc = __builtin_amdgcn_wmma_f32_16x16x32_bf16(false, a, false, b,
                                                      (short)0, acc, false, false);
    }
    const int lane = threadIdx.x & 31;
    const int n    = tn * 16 + (lane & 15);
    const int mb   = tm * 16 + ((lane >> 4) << 3);
    const float bv = bias ? bias[n] : 0.0f;
#pragma unroll
    for (int j = 0; j < 8; ++j)
        C[(long)(mb + j) * ldc + n] = acc[j] + bv;
}

// ---------------------------------------------------------------------------
// Fused LSTM step.  One block = 256 threads = 8 waves covers ALL 128 batch
// rows for one 16-wide hidden slice (ht) across the four gates, so each W_hh
// element is DMA'd into LDS exactly once per step and reused by 8 waves.
// K is processed in 8 chunks of 128, TDM double-buffered (issue chunk k+1,
// s_wait_tensorcnt 4 -> chunk k complete, consume, barrier, repeat).
// grid = 64 blocks (H/16), block = 256 threads.
// ---------------------------------------------------------------------------
__global__ void lstm_step_fused(const __bf16* __restrict__ h_in,   // [B,H] bf16
                                const __bf16* __restrict__ Whh,    // [4H,H] bf16
                                const float*  __restrict__ z,      // [B,4H] f32
                                float*        __restrict__ c,      // [B,H] f32 (in/out)
                                __bf16*       __restrict__ h_out,  // [B,H] bf16
                                float*        __restrict__ h_f32)  // [B,H] f32
{
    constexpr int H  = 1024;
    constexpr int KC = 128;                         // K chunk
    __shared__ __bf16 smem[2][4][16][KC];           // 32 KB, double buffered

    const int wave = threadIdx.x >> 5;
    const int ht   = blockIdx.x;                    // hidden tile 0..63
    const int bt   = wave;                          // batch tile 0..7

    // Issue the 4 gate tiles of one K-chunk into LDS buffer `buf`.
    auto issue_chunk = [&](int kc, int buf) {
#pragma unroll
        for (int g = 0; g < 4; ++g)
            tdm_load_tile_16x128(&smem[buf][g][0][0],
                                 Whh + ((long)(g * H + ht * 16)) * H + kc * KC);
    };

    if (wave == 0) issue_chunk(0, 0);

    v8f acc[4] = {};
    const __bf16* Arow = h_in + (long)bt * 16 * H;

    for (int kc = 0; kc < H / KC; ++kc) {
        const int buf = kc & 1;
        if (wave == 0) {
            if (kc < H / KC - 1) {
                issue_chunk(kc + 1, buf ^ 1);       // prefetch next chunk
                __builtin_amdgcn_s_wait_tensorcnt(4); // current chunk landed
            } else {
                __builtin_amdgcn_s_wait_tensorcnt(0);
            }
        }
        __syncthreads();
#pragma unroll
        for (int kk = 0; kk < KC; kk += 32) {
            v16bf a = load_a_frag(Arow + kc * KC + kk, H);
#pragma unroll
            for (int g = 0; g < 4; ++g) {           // torch gate order i,f,g,o
                v16bf b = load_b_frag(&smem[buf][g][0][0] + kk, KC);
                acc[g] = __builtin_amdgcn_wmma_f32_16x16x32_bf16(
                             false, a, false, b, (short)0, acc[g], false, false);
            }
        }
        __syncthreads();                            // buffer reusable
    }

    // LSTM cell pointwise: D layout lanes 0-15 -> M=j,N=lane; 16-31 -> M=j+8.
    const int lane = threadIdx.x & 31;
    const int n    = ht * 16 + (lane & 15);
    const int mb   = bt * 16 + ((lane >> 4) << 3);
#pragma unroll
    for (int j = 0; j < 8; ++j) {
        const int  brow = mb + j;
        const long zb   = (long)brow * (4 * H);
        float iv = sigmoidf (acc[0][j] + z[zb +           n]);
        float fv = sigmoidf (acc[1][j] + z[zb +     H +   n]);
        float gv = tanh_fast(acc[2][j] + z[zb + 2 * H +   n]);
        float ov = sigmoidf (acc[3][j] + z[zb + 3 * H +   n]);
        const long ci = (long)brow * H + n;
        float cn = fv * c[ci] + iv * gv;
        c[ci] = cn;
        float hn = ov * tanh_fast(cn);
        h_out[ci] = (__bf16)hn;
        h_f32[ci] = hn;
    }
}

// ---- small helpers --------------------------------------------------------
__global__ void f32_to_bf16_k(const float* __restrict__ in, __bf16* __restrict__ out, int n) {
    int i = blockIdx.x * blockDim.x + threadIdx.x;
    if (i < n) out[i] = (__bf16)in[i];
}
__global__ void bias_sum_k(const float* a, const float* b, float* o, int n) {
    int i = blockIdx.x * blockDim.x + threadIdx.x;
    if (i < n) o[i] = a[i] + b[i];
}
__global__ void zero_init_k(__bf16* h0, float* c, int n) {
    int i = blockIdx.x * blockDim.x + threadIdx.x;
    if (i < n) { h0[i] = (__bf16)0.0f; c[i] = 0.0f; }
}
__global__ void final_copy_k(const float* __restrict__ h_f32, float* __restrict__ out,
                             int H, int B) {
    int i = blockIdx.x * blockDim.x + threadIdx.x;
    if (i < H) out[i] = h_f32[(long)(B - 1) * H + i];
}

// ---------------------------------------------------------------------------
extern "C" void kernel_launch(void* const* d_in, const int* in_sizes, int n_in,
                              void* d_out, int out_size, void* d_ws, size_t ws_size,
                              hipStream_t stream) {
    (void)in_sizes; (void)n_in; (void)out_size; (void)ws_size;
    constexpr int B = 128, E = 512, H = 1024, T = 256, O = 512;  // matches reference

    const float* x    = (const float*)d_in[0];
    const float* W_ih = (const float*)d_in[1];
    const float* W_hh = (const float*)d_in[2];
    const float* b_ih = (const float*)d_in[3];
    const float* b_hh = (const float*)d_in[4];
    const float* fc_w = (const float*)d_in[5];
    const float* fc_b = (const float*)d_in[6];
    float* out = (float*)d_out;

    // workspace carve-out (~17.4 MB total), 256B aligned slices
    char* wp = (char*)d_ws;
    auto carve = [&](size_t bytes) { void* p = wp; wp += (bytes + 255) & ~(size_t)255; return p; };
    __bf16* Whh_bf = (__bf16*)carve((size_t)4 * H * H * 2);   // 8.39 MB
    __bf16* Wih_bf = (__bf16*)carve((size_t)4 * H * E * 2);   // 4.19 MB
    __bf16* x_bf   = (__bf16*)carve((size_t)B * E * 2);
    __bf16* fcw_bf = (__bf16*)carve((size_t)O * H * 2);
    float*  zbuf   = (float*) carve((size_t)B * 4 * H * 4);   // 2.10 MB
    float*  bsum   = (float*) carve((size_t)4 * H * 4);
    float*  cbuf   = (float*) carve((size_t)B * H * 4);
    float*  hf32   = (float*) carve((size_t)B * H * 4);
    __bf16* hA     = (__bf16*)carve((size_t)B * H * 2);
    __bf16* hB     = (__bf16*)carve((size_t)B * H * 2);

    const int thr = 256;
    f32_to_bf16_k<<<(4 * H * H + thr - 1) / thr, thr, 0, stream>>>(W_hh, Whh_bf, 4 * H * H);
    f32_to_bf16_k<<<(4 * H * E + thr - 1) / thr, thr, 0, stream>>>(W_ih, Wih_bf, 4 * H * E);
    f32_to_bf16_k<<<(B * E     + thr - 1) / thr, thr, 0, stream>>>(x,    x_bf,   B * E);
    f32_to_bf16_k<<<(O * H     + thr - 1) / thr, thr, 0, stream>>>(fc_w, fcw_bf, O * H);
    bias_sum_k   <<<(4 * H     + thr - 1) / thr, thr, 0, stream>>>(b_ih, b_hh, bsum, 4 * H);
    zero_init_k  <<<(B * H     + thr - 1) / thr, thr, 0, stream>>>(hA, cbuf, B * H);

    // z = x @ W_ih^T + (b_ih + b_hh)   [B, 4H]
    wmma_gemm_bias<<<(B / 16) * (4 * H / 16) / 4, 128, 0, stream>>>(
        x_bf, Wih_bf, bsum, zbuf, B, 4 * H, E, (long)4 * H);

    __bf16* hin = hA; __bf16* hout = hB;
    for (int t = 0; t < T; ++t) {
        lstm_step_fused<<<H / 16, 256, 0, stream>>>(hin, Whh_bf, zbuf, cbuf, hout, hf32);
        // out_sequence[:, t, :] = h_t @ fc_w^T + fc_b   (row stride T*O)
        wmma_gemm_bias<<<(B / 16) * (O / 16) / 4, 128, 0, stream>>>(
            hout, fcw_bf, fc_b, out + (long)t * O, B, O, H, (long)T * O);
        __bf16* tmp = hin; hin = hout; hout = tmp;
    }

    // next_state = h_T[last batch row]
    final_copy_k<<<(H + thr - 1) / thr, thr, 0, stream>>>(hf32, out + (long)B * T * O, H, B);
}
